// MedDisSeg2_23811298690092
// MI455X (gfx1250) — compile-verified, matrix operand order checked by reading, not run on hardware
//
#include <hip/hip_runtime.h>
#include <hip/hip_bf16.h>

typedef _Float16 half_t;
typedef half_t v16h __attribute__((ext_vector_type(16)));
typedef half_t v8h  __attribute__((ext_vector_type(8)));
typedef float  v8f  __attribute__((ext_vector_type(8)));

#define HH 128
#define WW 128
#define DIM 128
#define BB 2
#define LL (HH * WW)
#define MM (BB * LL)
#define K4H 324
#define K4H_PAD 384
#define BN_EPS 1e-5f

union frag_u { v16h f; v8h h[2]; };

__device__ __forceinline__ v8h zero8() {
    v8h z = {(half_t)0.0f, (half_t)0.0f, (half_t)0.0f, (half_t)0.0f,
             (half_t)0.0f, (half_t)0.0f, (half_t)0.0f, (half_t)0.0f};
    return z;
}

// ---------------- gfx1250 async global->LDS copy (with safe fallback) ----------------
#if __has_builtin(__builtin_amdgcn_global_load_async_to_lds_b128) && \
    __has_builtin(__builtin_amdgcn_s_wait_asynccnt)
#define HAVE_ASYNC_LDS 1
typedef int v4i_vs __attribute__((__vector_size__(4 * sizeof(int))));
__device__ __forceinline__ void copy16_g2lds(const half_t* g, half_t* l) {
    __builtin_amdgcn_global_load_async_to_lds_b128(
        (v4i_vs*)(unsigned long long)g, (v4i_vs*)(unsigned long long)l, 0, 0);
}
__device__ __forceinline__ void async_copy_wait() {
    __builtin_amdgcn_s_wait_asynccnt(0);
}
#else
#define HAVE_ASYNC_LDS 0
__device__ __forceinline__ void copy16_g2lds(const half_t* g, half_t* l) {
    *(v8h*)l = *(const v8h*)g;
}
__device__ __forceinline__ void async_copy_wait() {}
#endif

// ---------------- WMMA fragment helpers (ISA 7.12.2 layouts) ----------------
// A 16x32 f16: lane<16 -> M=lane; each lane's 16 halves are two contiguous
// 8-half runs in a row-major tile -> two 16B LDS loads.
__device__ __forceinline__ v16h load_a_frag(const half_t* base, int ldk) {
    const int lane = threadIdx.x & 31;
    const int m    = lane & 15;
    const int hf   = lane >> 4;
    const half_t* row = base + m * ldk + 8 * hf;
    frag_u u;
    u.h[0] = *(const v8h*)(row);
    u.h[1] = *(const v8h*)(row + 16);
    return u.f;
}

// B fragments pre-swizzled into fragment-ready tiles: [32 lanes][16 halves],
// one 32B contiguous load per lane.
__device__ __forceinline__ v16h load_b_frag_sw(const half_t* tile) {
    const int lane = threadIdx.x & 31;
    return *(const v16h*)(tile + lane * 16);
}

__device__ __forceinline__ v8f wmma_f16(v16h a, v16h b, v8f c) {
    return __builtin_amdgcn_wmma_f32_16x16x32_f16(false, a, false, b, (short)0, c, false, false);
}

__device__ __forceinline__ float telu(float y) { return y * tanhf(__expf(y)); }

// decode fragment element -> logical (k_local, n_local) of a 32x16 B tile
__device__ __forceinline__ void b_frag_decode(int lane, int e, int& kl, int& nl) {
    int hf = lane >> 4;
    int v  = e >> 1;
    int pr = e & 1;
    kl = (v < 4 ? 2 * v : 8 + 2 * v) + 8 * hf + pr;
    nl = lane & 15;
}

// ---------------- layout / prep kernels ----------------

// NCHW f32 -> (B*HW, C) f16 channels-last
__global__ void k_nchw_to_nlc(const float* __restrict__ in, half_t* __restrict__ out,
                              int C, int total) {
    int idx = blockIdx.x * 256 + threadIdx.x;
    if (idx >= total) return;
    int hw  = LL;
    int b   = idx / (C * hw);
    int r   = idx % (C * hw);
    int c   = r / hw;
    int p   = r % hw;
    out[((long long)b * hw + p) * C + c] = (half_t)in[idx];
}

// OIHW f32 (128, Cin, 3, 3) -> swizzled f16 tiles [9*Cin/32][8][32][16]
// logical B[k][o], k = t*Cin + c (tap-major, channel-minor)
__global__ void k_prep_conv_w_sw(const float* __restrict__ w, half_t* __restrict__ out,
                                 int Cin) {
    int idx = blockIdx.x * 256 + threadIdx.x;
    int total = 9 * Cin * DIM;
    if (idx >= total) return;
    int tile = idx >> 9;
    int e16  = idx & 511;
    int lane = e16 >> 4;
    int e    = e16 & 15;
    int kt = tile >> 3;
    int nt = tile & 7;
    int kl, nl;
    b_frag_decode(lane, e, kl, nl);
    int k = kt * 32 + kl;
    int o = nt * 16 + nl;
    int t = k / Cin;
    int c = k % Cin;
    out[idx] = (half_t)w[(o * Cin + c) * 9 + t];
}

// (N, K=128) f32 -> swizzled f16 tiles [4][Npad/16][32][16], zero padded cols
__global__ void k_prep_lin_w_sw(const float* __restrict__ w, half_t* __restrict__ out,
                                int N, int Npad) {
    int idx = blockIdx.x * 256 + threadIdx.x;
    int total = DIM * Npad;
    if (idx >= total) return;
    int tile = idx >> 9;
    int e16  = idx & 511;
    int lane = e16 >> 4;
    int e    = e16 & 15;
    int ntiles = Npad >> 4;
    int kt = tile / ntiles;
    int nt = tile % ntiles;
    int kl, nl;
    b_frag_decode(lane, e, kl, nl);
    int k = kt * 32 + kl;
    int n = nt * 16 + nl;
    out[idx] = (n < N) ? (half_t)w[n * DIM + k] : (half_t)0.0f;
}

// ---------------- Conv3x3 + BN + TeLU (implicit GEMM, WMMA) ----------------
// block = 128 threads = 4 waves; each wave: 16 pixels x 128 out-channels.
// grid = B*H*2 (64-pixel tiles along a row). A staged in LDS with halo (f16 in, f16 out).
template <int CIN>
__global__ void k_conv3x3_bn_telu(const half_t* __restrict__ in_cl,
                                  const half_t* __restrict__ Wsw,   // swizzled tiles
                                  const float* __restrict__ gamma,
                                  const float* __restrict__ beta,
                                  const float* __restrict__ mean,
                                  const float* __restrict__ var,
                                  void* __restrict__ out_v, int store_nchw_f32) {
    constexpr int LDC = CIN + 8;
    constexpr int CV  = CIN / 8;
    __shared__ half_t s[3 * 66 * LDC];

    const int blk = blockIdx.x;
    const int xb  = blk & 1;
    const int y   = (blk >> 1) & (HH - 1);
    const int b   = blk >> 8;
    const int x0  = xb * 64;

    const int totalv = 3 * 66 * CV;
    for (int i = threadIdx.x; i < totalv; i += 128) {
        int rr  = i / (66 * CV);
        int rem = i % (66 * CV);
        int col = rem / CV;
        int cv  = rem % CV;
        int yy  = y + rr - 1;
        int xx  = x0 + col - 1;
        half_t* dst = &s[(rr * 66 + col) * LDC + cv * 8];
        if (yy >= 0 && yy < HH && xx >= 0 && xx < WW)
            copy16_g2lds(&in_cl[(((long long)b * HH + yy) * WW + xx) * CIN + cv * 8], dst);
        else
            *(v8h*)dst = zero8();
    }
    async_copy_wait();
    __syncthreads();

    const int wave = threadIdx.x >> 5;
    const int px0  = wave * 16;

    v8f acc[8];
    const v8f vzero = {0.f, 0.f, 0.f, 0.f, 0.f, 0.f, 0.f, 0.f};
#pragma unroll
    for (int i = 0; i < 8; ++i) acc[i] = vzero;

#pragma unroll
    for (int t = 0; t < 9; ++t) {
        const int di = t / 3, dj = t % 3;
#pragma unroll
        for (int kc = 0; kc < CIN / 32; ++kc) {
            const int kt = t * (CIN / 32) + kc;  // global K-tile index
            if (kt + 1 < 9 * (CIN / 32))
                __builtin_prefetch(Wsw + ((size_t)(kt + 1) * 8) * 512, 0, 3);
            const half_t* abase = &s[(di * 66 + px0 + dj) * LDC + kc * 32];
            v16h afrag = load_a_frag(abase, LDC);
#pragma unroll
            for (int nt = 0; nt < 8; ++nt) {
                v16h bfrag = load_b_frag_sw(Wsw + ((size_t)kt * 8 + nt) * 512);
                acc[nt] = wmma_f16(afrag, bfrag, acc[nt]);
            }
        }
    }

    const int lane = threadIdx.x & 31;
    const int nn   = lane & 15;
    const int hf   = lane >> 4;
#pragma unroll
    for (int nt = 0; nt < 8; ++nt) {
        int n = nt * 16 + nn;
        float sc = gamma[n] * rsqrtf(var[n] + BN_EPS);
        float mu = mean[n];
        float bt = beta[n];
#pragma unroll
        for (int r = 0; r < 8; ++r) {
            float yv = (acc[nt][r] - mu) * sc + bt;
            float o  = telu(yv);
            int x = x0 + px0 + r + 8 * hf;
            if (store_nchw_f32)
                ((float*)out_v)[(((long long)b * DIM + n) * HH + y) * WW + x] = o;
            else
                ((half_t*)out_v)[(((long long)b * HH + y) * WW + x) * DIM + n] = (half_t)o;
        }
    }
}

// ---------------- generic K=128 GEMM: out = (A @ Bw + bias) * scale ----------------
// block = 256 threads = 8 waves (each: 16 rows x 128 cols of this n-block)
// grid = (M/128, Npad/128). A is f16 channels-last; out f16 or f32.
__global__ void k_gemm128(const half_t* __restrict__ A,     // (M,128) f16
                          const half_t* __restrict__ Bsw,   // swizzled [4][Npad/16][512]
                          const float* __restrict__ bias,   // [Nout] or nullptr
                          void* __restrict__ out_v,         // (M, Nout)
                          int Npad, int Nout, float scale, int out_f16) {
    __shared__ half_t sA[128 * 136];

    const int row0 = blockIdx.x * 128;
    const int nb0  = blockIdx.y * 128;
    const int ntiles = Npad >> 4;

    for (int i = threadIdx.x; i < 128 * 16; i += 256) {
        int m = i >> 4, kv = i & 15;
        copy16_g2lds(&A[((long long)row0 + m) * DIM + kv * 8], &sA[m * 136 + kv * 8]);
    }
    async_copy_wait();
    __syncthreads();

    const int wave  = threadIdx.x >> 5;
    const int mbase = wave * 16;

    v8f acc[8];
    const v8f vzero = {0.f, 0.f, 0.f, 0.f, 0.f, 0.f, 0.f, 0.f};
#pragma unroll
    for (int i = 0; i < 8; ++i) acc[i] = vzero;

#pragma unroll
    for (int kc = 0; kc < 4; ++kc) {
        if (kc < 3)
            __builtin_prefetch(Bsw + ((size_t)(kc + 1) * ntiles + (nb0 >> 4)) * 512, 0, 3);
        v16h afrag = load_a_frag(&sA[mbase * 136 + kc * 32], 136);
#pragma unroll
        for (int nt = 0; nt < 8; ++nt) {
            v16h bfrag = load_b_frag_sw(
                Bsw + ((size_t)kc * ntiles + (nb0 >> 4) + nt) * 512);
            acc[nt] = wmma_f16(afrag, bfrag, acc[nt]);
        }
    }

    const int lane = threadIdx.x & 31;
    const int nn   = lane & 15;
    const int hf   = lane >> 4;
#pragma unroll
    for (int nt = 0; nt < 8; ++nt) {
        int n = nb0 + nt * 16 + nn;
        if (n >= Nout) continue;
        float bv = bias ? bias[n] : 0.0f;
#pragma unroll
        for (int r = 0; r < 8; ++r) {
            int m = mbase + r + 8 * hf;
            float val = (acc[nt][r] + bv) * scale;
            long long oi = ((long long)row0 + m) * Nout + n;
            if (out_f16) ((half_t*)out_v)[oi] = (half_t)val;
            else         ((float*)out_v)[oi]  = val;
        }
    }
}

// ---------------- softmax over last dim 9 of (M, 4, 9, 9), f32 in place ----------------
__global__ void k_softmax9(float* __restrict__ a, int rows) {
    int r = blockIdx.x * 256 + threadIdx.x;
    if (r >= rows) return;
    int pixel = r / 36;
    int w     = r % 36;
    float* p  = a + (long long)pixel * K4H + w * 9;
    float mx = p[0];
#pragma unroll
    for (int i = 1; i < 9; ++i) mx = fmaxf(mx, p[i]);
    float e[9], s = 0.0f;
#pragma unroll
    for (int i = 0; i < 9; ++i) { e[i] = __expf(p[i] - mx); s += e[i]; }
    float inv = 1.0f / s;
#pragma unroll
    for (int i = 0; i < 9; ++i) p[i] = e[i] * inv;
}

// ---------------- fused attention apply + fold (gather form) ----------------
// out[y,x,c] = sum_{p=(i,j)} sum_{q=(qi,qj)} attn[l'=(y-i+1,x-j+1)][h][p][q] *
//              v[(y+qi-i, x+qj-j), c],   h = c/32
// grid = B*L blocks, 128 threads (one channel each); v is f16, attn f32, out f16
__global__ void k_attn_apply_fold(const float* __restrict__ attn,   // (M, 324)
                                  const half_t* __restrict__ v,     // (M, 128) f16
                                  half_t* __restrict__ out) {       // (M, 128) f16
    __shared__ float s_a[K4H];
    __shared__ float s_v[25 * 128];

    const int pix = blockIdx.x;
    const int b   = pix >> 14;
    const int p   = pix & (LL - 1);
    const int y   = p >> 7;
    const int x   = p & (WW - 1);

    for (int i = threadIdx.x; i < K4H; i += 128) {
        int t   = i / 36;
        int rem = i % 36;
        int h   = rem / 9;
        int q   = rem % 9;
        int di = t / 3, dj = t % 3;
        int yy = y - di + 1, xx = x - dj + 1;
        float av = 0.0f;
        if (yy >= 0 && yy < HH && xx >= 0 && xx < WW) {
            long long lp = ((long long)b * HH + yy) * WW + xx;
            av = attn[lp * K4H + h * 81 + t * 9 + q];
        }
        s_a[i] = av;
    }
    for (int i = threadIdx.x; i < 25 * 16; i += 128) {
        int pos = i >> 4;
        int cv  = i & 15;
        int dy = pos / 5 - 2, dx = pos % 5 - 2;
        int yy = y + dy, xx = x + dx;
        v8h vv = zero8();
        if (yy >= 0 && yy < HH && xx >= 0 && xx < WW)
            vv = *(const v8h*)&v[(((long long)b * HH + yy) * WW + xx) * 128 + cv * 8];
#pragma unroll
        for (int j = 0; j < 8; ++j) s_v[pos * 128 + cv * 8 + j] = (float)vv[j];
    }
    __syncthreads();

    const int c = threadIdx.x;
    const int h = c >> 5;
    float sum = 0.0f;
#pragma unroll
    for (int t = 0; t < 9; ++t) {
        int di = t / 3, dj = t % 3;
#pragma unroll
        for (int q = 0; q < 9; ++q) {
            int qi = q / 3, qj = q % 3;
            int pos = (qi - di + 2) * 5 + (qj - dj + 2);
            sum += s_a[t * 36 + h * 9 + q] * s_v[pos * 128 + c];
        }
    }
    out[(long long)pix * 128 + c] = (half_t)sum;
}

// ---------------- host orchestration ----------------

extern "C" void kernel_launch(void* const* d_in, const int* in_sizes, int n_in,
                              void* d_out, int out_size, void* d_ws, size_t ws_size,
                              hipStream_t stream) {
    (void)in_sizes; (void)n_in; (void)out_size; (void)ws_size;

    const float* x      = (const float*)d_in[0];
    const float* fg     = (const float*)d_in[1];
    const float* bg     = (const float*)d_in[2];
    const float* w_in1  = (const float*)d_in[3];
    const float* w_in2  = (const float*)d_in[4];
    const float* w_out1 = (const float*)d_in[5];
    const float* w_out2 = (const float*)d_in[6];
    const float* bn_g   = (const float*)d_in[7];
    const float* bn_b   = (const float*)d_in[8];
    const float* bn_m   = (const float*)d_in[9];
    const float* bn_v   = (const float*)d_in[10];
    const float* v_w    = (const float*)d_in[11];
    const float* v_b    = (const float*)d_in[12];
    const float* afg_w  = (const float*)d_in[13];
    const float* afg_b  = (const float*)d_in[14];
    const float* abg_w  = (const float*)d_in[15];
    const float* abg_b  = (const float*)d_in[16];
    const float* proj_w = (const float*)d_in[17];
    const float* proj_b = (const float*)d_in[18];

    char* ws = (char*)d_ws;
    size_t off = 0;
    auto alloc = [&](size_t bytes) -> void* {
        off = (off + 255) & ~(size_t)255;
        void* pp = ws + off;
        off += bytes;
        return pp;
    };

    half_t* x_cl    = (half_t*)alloc((size_t)MM * 32 * 2);
    half_t* featA   = (half_t*)alloc((size_t)MM * DIM * 2);
    half_t* featB   = (half_t*)alloc((size_t)MM * DIM * 2);
    half_t* v_cl    = (half_t*)alloc((size_t)MM * DIM * 2);
    half_t* fb_cl   = (half_t*)alloc((size_t)MM * DIM * 2);
    float*  logits  = (float*)alloc((size_t)MM * K4H * 4);
    half_t* foldbuf = (half_t*)alloc((size_t)MM * DIM * 2);
    half_t* xw_fg   = (half_t*)alloc((size_t)MM * DIM * 2);
    half_t* xw_bg   = (half_t*)alloc((size_t)MM * DIM * 2);
    half_t* wc1   = (half_t*)alloc((size_t)9 * 32 * DIM * 2);
    half_t* wc2   = (half_t*)alloc((size_t)9 * DIM * DIM * 2);
    half_t* wo1   = (half_t*)alloc((size_t)9 * DIM * DIM * 2);
    half_t* wo2   = (half_t*)alloc((size_t)9 * DIM * DIM * 2);
    half_t* wv    = (half_t*)alloc((size_t)DIM * DIM * 2);
    half_t* wproj = (half_t*)alloc((size_t)DIM * DIM * 2);
    half_t* wafg  = (half_t*)alloc((size_t)DIM * K4H_PAD * 2);
    half_t* wabg  = (half_t*)alloc((size_t)DIM * K4H_PAD * 2);

    const float scale = 0.17677669529663687f;  // (dim/heads)^-0.5 = 1/sqrt(32)

    // layout transforms
    {
        int tot = BB * 32 * LL;
        k_nchw_to_nlc<<<(tot + 255) / 256, 256, 0, stream>>>(x, x_cl, 32, tot);
    }
    {
        int tot = BB * DIM * LL;
        k_nchw_to_nlc<<<(tot + 255) / 256, 256, 0, stream>>>(fg, fb_cl, DIM, tot);
    }
    // weight repacks (fragment-ready swizzled tiles)
    k_prep_conv_w_sw<<<(DIM * 32 * 9 + 255) / 256, 256, 0, stream>>>(w_in1, wc1, 32);
    k_prep_conv_w_sw<<<(DIM * DIM * 9 + 255) / 256, 256, 0, stream>>>(w_in2, wc2, DIM);
    k_prep_conv_w_sw<<<(DIM * DIM * 9 + 255) / 256, 256, 0, stream>>>(w_out1, wo1, DIM);
    k_prep_conv_w_sw<<<(DIM * DIM * 9 + 255) / 256, 256, 0, stream>>>(w_out2, wo2, DIM);
    k_prep_lin_w_sw<<<(DIM * DIM + 255) / 256, 256, 0, stream>>>(v_w, wv, DIM, DIM);
    k_prep_lin_w_sw<<<(DIM * DIM + 255) / 256, 256, 0, stream>>>(proj_w, wproj, DIM, DIM);
    k_prep_lin_w_sw<<<(DIM * K4H_PAD + 255) / 256, 256, 0, stream>>>(afg_w, wafg, K4H, K4H_PAD);
    k_prep_lin_w_sw<<<(DIM * K4H_PAD + 255) / 256, 256, 0, stream>>>(abg_w, wabg, K4H, K4H_PAD);

    const int conv_grid = BB * HH * 2;
    // input_cbt
    k_conv3x3_bn_telu<32><<<conv_grid, 128, 0, stream>>>(
        x_cl, wc1, bn_g + 0 * DIM, bn_b + 0 * DIM, bn_m + 0 * DIM, bn_v + 0 * DIM, featA, 0);
    k_conv3x3_bn_telu<128><<<conv_grid, 128, 0, stream>>>(
        featA, wc2, bn_g + 1 * DIM, bn_b + 1 * DIM, bn_m + 1 * DIM, bn_v + 1 * DIM, featB, 0);

    // v projection
    k_gemm128<<<dim3(MM / 128, 1), 256, 0, stream>>>(featB, wv, v_b, v_cl, DIM, DIM, 1.0f, 1);

    // fg attention: logits -> softmax -> apply+fold -> proj
    k_gemm128<<<dim3(MM / 128, K4H_PAD / 128), 256, 0, stream>>>(
        fb_cl, wafg, afg_b, logits, K4H_PAD, K4H, scale, 0);
    k_softmax9<<<((MM * 36) + 255) / 256, 256, 0, stream>>>(logits, MM * 36);
    k_attn_apply_fold<<<MM, 128, 0, stream>>>(logits, v_cl, foldbuf);
    k_gemm128<<<dim3(MM / 128, 1), 256, 0, stream>>>(foldbuf, wproj, proj_b, xw_fg, DIM, DIM, 1.0f, 1);

    // bg attention on fg-weighted features
    {
        int tot = BB * DIM * LL;
        k_nchw_to_nlc<<<(tot + 255) / 256, 256, 0, stream>>>(bg, fb_cl, DIM, tot);
    }
    k_gemm128<<<dim3(MM / 128, K4H_PAD / 128), 256, 0, stream>>>(
        fb_cl, wabg, abg_b, logits, K4H_PAD, K4H, scale, 0);
    k_softmax9<<<((MM * 36) + 255) / 256, 256, 0, stream>>>(logits, MM * 36);
    k_attn_apply_fold<<<MM, 128, 0, stream>>>(logits, xw_fg, foldbuf);
    k_gemm128<<<dim3(MM / 128, 1), 256, 0, stream>>>(foldbuf, wproj, proj_b, xw_bg, DIM, DIM, 1.0f, 1);

    // output_cbt (second conv writes f32 NCHW straight to d_out)
    k_conv3x3_bn_telu<128><<<conv_grid, 128, 0, stream>>>(
        xw_bg, wo1, bn_g + 2 * DIM, bn_b + 2 * DIM, bn_m + 2 * DIM, bn_v + 2 * DIM, featA, 0);
    k_conv3x3_bn_telu<128><<<conv_grid, 128, 0, stream>>>(
        featA, wo2, bn_g + 3 * DIM, bn_b + 3 * DIM, bn_m + 3 * DIM, bn_v + 3 * DIM,
        d_out, 1);
}
